// X2CTSwinTransformer_50989851738178
// MI455X (gfx1250) — compile-verified, matrix operand order checked by reading, not run on hardware
//
#include <hip/hip_runtime.h>

// ---------------- static config ----------------
#define TOK   262144          // total tokens = 64*64*64
#define CCH   96              // channels
#define NHD   3               // heads
#define HD    32              // head dim
#define NWIN  4096            // windows
#define NTOK  64              // tokens per window
#define SSH   2               // shift
#define SCALE 0.17677669529663687f
#define MSK   (-100.0f)

typedef __bf16 v16bf __attribute__((ext_vector_type(16)));
typedef float  v8f   __attribute__((ext_vector_type(8)));
typedef unsigned int u32x4 __attribute__((ext_vector_type(4)));
typedef int          i32x8 __attribute__((ext_vector_type(8)));
typedef int          i32x4 __attribute__((ext_vector_type(4)));

struct __align__(16) U128 { unsigned int a, b, c, d; };
union V16U {
    v16bf v;
    U128 q[2];
    unsigned short u[16];
};

static __device__ __forceinline__ unsigned short f2bf(float f) {
    unsigned int u = __float_as_uint(f);
    u = (u + 0x7FFFu + ((u >> 16) & 1u)) >> 16;
    return (unsigned short)u;
}

static __device__ __forceinline__ v8f wmma_bf16(v16bf a, v16bf b, v8f c) {
    return __builtin_amdgcn_wmma_f32_16x16x32_bf16(false, a, false, b, (short)0, c, false, false);
}

// A-tile (16xK32 bf16) load, row-major source (global or LDS). ISA 7.12.2 layout.
static __device__ __forceinline__ v16bf load_tile_bf(const unsigned short* base, int row0,
                                                     int ld, int k0, int lane) {
    int m  = lane & 15;
    int kb = (lane >> 4) << 3;
    const unsigned short* p = base + (size_t)(row0 + m) * ld + k0 + kb;
    V16U r;
    r.q[0] = *reinterpret_cast<const U128*>(p);
    r.q[1] = *reinterpret_cast<const U128*>(p + 16);
    return r.v;
}

// TDM: async-load a rows x cols (bf16) row-major matrix from global into LDS.
// D# per CDNA5 ISA ch.8: group0 = {count=1, lds_addr, global_addr, type=2},
// group1 = {data_size=1(2B), tensor_dim0/1, tile_dim0/1, dim0_stride}.
static __device__ __forceinline__ void tdm_load_2d(unsigned lds_off, const void* gaddr,
                                                   unsigned rows, unsigned cols) {
    unsigned long long ga = (unsigned long long)(uintptr_t)gaddr;
    u32x4 g0;
    g0[0] = 1u;                                              // count=1 (valid), user mode
    g0[1] = lds_off;                                         // lds_addr (bytes)
    g0[2] = (unsigned)(ga & 0xFFFFFFFFu);                    // global_addr[31:0]
    g0[3] = (unsigned)((ga >> 32) & 0x01FFFFFFu) | (2u << 30); // global_addr[56:32] | type=2
    i32x8 g1;
    g1[0] = (int)(1u << 16);                                 // data_size = 1 -> 2 bytes
    g1[1] = (int)((cols & 0xFFFFu) << 16);                   // tensor_dim0[15:0] @ bit48
    g1[2] = (int)((cols >> 16) | ((rows & 0xFFFFu) << 16));  // tensor_dim0[31:16] | tensor_dim1[15:0]
    g1[3] = (int)((rows >> 16) | ((cols & 0xFFFFu) << 16));  // tensor_dim1[31:16] | tile_dim0
    g1[4] = (int)(rows & 0xFFFFu);                           // tile_dim1 (tile_dim2 = 0)
    g1[5] = (int)cols;                                       // tensor_dim0_stride[31:0]
    g1[6] = 0;
    g1[7] = 0;
    i32x4 z4 = {0, 0, 0, 0};
    i32x8 z8 = {0, 0, 0, 0, 0, 0, 0, 0};
    __builtin_amdgcn_tensor_load_to_lds(g0, g1, z4, z4, z8, 0);
}

static __device__ __forceinline__ unsigned lds_offset_of(const void* p) {
    return (unsigned)(uintptr_t)p;   // low 32 bits of generic ptr = LDS byte offset
}

// windowed token index -> global flat token index (shift composed with partition)
static __device__ __forceinline__ size_t win_to_global(int o) {
    int win = o >> 6, n = o & 63;
    int zw = win >> 8, hw = (win >> 4) & 15, ww = win & 15;
    int dz = n >> 4, dh = (n >> 2) & 3, dw = n & 3;
    int gz = (zw * 4 + dz + SSH) & 63;
    int gh = (hw * 4 + dh + SSH) & 63;
    int gw = (ww * 4 + dw + SSH) & 63;
    return ((size_t)gz * 64 + gh) * 64 + gw;
}

static __device__ __forceinline__ int regio(int u) { return (u < 60) ? 0 : ((u < 62) ? 1 : 2); }

// ---------------- K0: fp32 -> bf16 weight pre-conversion ----------------
__global__ __launch_bounds__(256) void k_cvt(const float* __restrict__ src,
                                             unsigned short* __restrict__ dst, int n) {
    int i = blockIdx.x * 256 + threadIdx.x;
    if (i < n) dst[i] = f2bf(src[i]);
}

// ---------------- K1: LN1 + shift + window partition -> bf16 ----------------
__global__ __launch_bounds__(256) void k_ln1_win(const float* __restrict__ x,
                                                 const float* __restrict__ g,
                                                 const float* __restrict__ b,
                                                 unsigned short* __restrict__ xw) {
    int o    = (blockIdx.x * 256 + threadIdx.x) >> 5;   // windowed token
    int lane = threadIdx.x & 31;
    if (o >= TOK) return;
    size_t t = win_to_global(o);
    const float* row = x + t * CCH;
    float v0 = row[lane], v1 = row[lane + 32], v2 = row[lane + 64];
    float s = v0 + v1 + v2;
#pragma unroll
    for (int off = 16; off; off >>= 1) s += __shfl_xor(s, off);
    float mean = s * (1.0f / 96.0f);
    float d0 = v0 - mean, d1 = v1 - mean, d2 = v2 - mean;
    float q = d0 * d0 + d1 * d1 + d2 * d2;
#pragma unroll
    for (int off = 16; off; off >>= 1) q += __shfl_xor(q, off);
    float rstd = rsqrtf(q * (1.0f / 96.0f) + 1e-5f);
    unsigned short* out = xw + (size_t)o * CCH;
    out[lane]      = f2bf(d0 * rstd * g[lane]      + b[lane]);
    out[lane + 32] = f2bf(d1 * rstd * g[lane + 32] + b[lane + 32]);
    out[lane + 64] = f2bf(d2 * rstd * g[lane + 64] + b[lane + 64]);
}

// ---------------- K2: QKV GEMM, W staged in LDS via TDM ----------------
__global__ __launch_bounds__(256) void k_qkv(const unsigned short* __restrict__ xw,
                                             const unsigned short* __restrict__ W,
                                             const float* __restrict__ bias,
                                             unsigned short* __restrict__ qkv) {
    __shared__ __align__(16) unsigned short s_w[288 * 96];   // 55296 B
    if ((threadIdx.x >> 5) == 0) {                           // wave 0 issues the DMA
        tdm_load_2d(lds_offset_of(s_w), W, 288, 96);
        __builtin_amdgcn_s_wait_tensorcnt(0);
    }
    __syncthreads();

    int mt   = (blockIdx.x * 256 + threadIdx.x) >> 5;        // M-tile (16 tokens)
    int lane = threadIdx.x & 31;
    if (mt >= TOK / 16) return;
    v16bf A[3];
#pragma unroll
    for (int kc = 0; kc < 3; kc++) A[kc] = load_tile_bf(xw, mt * 16, CCH, kc * 32, lane);
    int colL = lane & 15, mhi = (lane >> 4) << 3;
#pragma unroll
    for (int nt = 0; nt < 18; nt++) {
        v8f acc = {};
#pragma unroll
        for (int kc = 0; kc < 3; kc++)
            acc = wmma_bf16(A[kc], load_tile_bf(s_w, nt * 16, CCH, kc * 32, lane), acc);
        float bi = bias[nt * 16 + colL];
        int part = nt / 6, sub = nt % 6;
        int head = sub >> 1;
        int d    = ((sub & 1) << 4) + colL;
        float sc = (part == 0) ? SCALE : 1.0f;
#pragma unroll
        for (int r = 0; r < 8; r++) {
            int row = mt * 16 + r + mhi;
            int win = row >> 6, n = row & 63;
            size_t off = ((((size_t)win * 3 + part) * NHD + head) * NTOK + n) * HD + d;
            qkv[off] = f2bf((acc[r] + bi) * sc);
        }
    }
}

// ---------------- K3: windowed attention, one head per wave ----------------
__global__ __launch_bounds__(96) void k_attn(const unsigned short* __restrict__ qkv,
                                             const float* __restrict__ rel_bias,
                                             unsigned short* __restrict__ outw) {
    __shared__ float s_rel[343 * NHD];
    __shared__ float s_P[NHD][16 * 64];
    int tid = threadIdx.x, lane = tid & 31, hv = tid >> 5;   // hv = head
    int win = blockIdx.x;
    for (int i = tid; i < 343 * NHD; i += 96) s_rel[i] = rel_bias[i];
    __syncthreads();

    const unsigned short* qb = qkv + ((((size_t)win * 3 + 0) * NHD + hv) * NTOK) * HD;
    const unsigned short* kb = qkv + ((((size_t)win * 3 + 1) * NHD + hv) * NTOK) * HD;
    const unsigned short* vb = qkv + ((((size_t)win * 3 + 2) * NHD + hv) * NTOK) * HD;

    // K as B-tiles (scores): key token = B column, head-dim = contiguous K
    v16bf Bk[4];
#pragma unroll
    for (int t = 0; t < 4; t++) Bk[t] = load_tile_bf(kb, t * 16, HD, 0, lane);

    // V as B-tiles (PV): column = head dim, K = key token (scattered)
    v16bf Bv[2][2];
    {
        int nn = lane & 15, kb8 = (lane >> 4) << 3;
#pragma unroll
        for (int kc = 0; kc < 2; kc++)
#pragma unroll
            for (int ntl = 0; ntl < 2; ntl++) {
                V16U r;
#pragma unroll
                for (int e = 0; e < 8; e++)
                    r.u[e] = vb[(size_t)(kc * 32 + kb8 + e) * HD + ntl * 16 + nn];
#pragma unroll
                for (int e = 0; e < 8; e++)
                    r.u[8 + e] = vb[(size_t)(kc * 32 + 16 + kb8 + e) * HD + ntl * 16 + nn];
                Bv[kc][ntl] = r.v;
            }
    }

    int zw = win >> 8, hw = (win >> 4) & 15, ww = win & 15;
    float* P = s_P[hv];
    int mlo  = (lane >= 16) ? 8 : 0;
    int colL = lane & 15;

    for (int mt = 0; mt < 4; mt++) {
        v16bf Aq = load_tile_bf(qb, mt * 16, HD, 0, lane);
        v8f S[4];
#pragma unroll
        for (int t = 0; t < 4; t++) { S[t] = {}; S[t] = wmma_bf16(Aq, Bk[t], S[t]); }

        // bias + shifted-window mask in C-layout, spill to LDS for softmax
#pragma unroll
        for (int t = 0; t < 4; t++) {
#pragma unroll
            for (int r = 0; r < 8; r++) {
                int nq = mt * 16 + r + mlo;        // query token
                int mk = t * 16 + colL;            // key token
                int dz1 = nq >> 4, dh1 = (nq >> 2) & 3, dw1 = nq & 3;
                int dz2 = mk >> 4, dh2 = (mk >> 2) & 3, dw2 = mk & 3;
                int idx = (dz1 - dz2 + 3) * 49 + (dh1 - dh2 + 3) * 7 + (dw1 - dw2 + 3);
                float v = S[t][r] + s_rel[idx * NHD + hv];
                int c1 = regio(zw * 4 + dz1) * 9 + regio(hw * 4 + dh1) * 3 + regio(ww * 4 + dw1);
                int c2 = regio(zw * 4 + dz2) * 9 + regio(hw * 4 + dh2) * 3 + regio(ww * 4 + dw2);
                if (c1 != c2) v += MSK;
                P[(r + mlo) * 64 + mk] = v;
            }
        }
        __syncthreads();

        if (lane < 16) {                           // one row per lane
            float* row = P + lane * 64;
            float mx = -1e30f;
            for (int j = 0; j < 64; j++) mx = fmaxf(mx, row[j]);
            float sum = 0.0f;
            for (int j = 0; j < 64; j++) { float e = __expf(row[j] - mx); row[j] = e; sum += e; }
            float inv = 1.0f / sum;
            for (int j = 0; j < 64; j++) row[j] *= inv;
        }
        __syncthreads();

        // P as A-tiles (K=64 -> 2 chunks), accumulate O = P @ V
        v8f O[2];
        O[0] = {}; O[1] = {};
        int m = lane & 15, kb8 = (lane >> 4) << 3;
#pragma unroll
        for (int kc = 0; kc < 2; kc++) {
            V16U pa;
#pragma unroll
            for (int e = 0; e < 8; e++) pa.u[e] = f2bf(P[m * 64 + kc * 32 + kb8 + e]);
#pragma unroll
            for (int e = 0; e < 8; e++) pa.u[8 + e] = f2bf(P[m * 64 + kc * 32 + 16 + kb8 + e]);
#pragma unroll
            for (int ntl = 0; ntl < 2; ntl++) O[ntl] = wmma_bf16(pa.v, Bv[kc][ntl], O[ntl]);
        }
        __syncthreads();   // P reused next iteration

#pragma unroll
        for (int ntl = 0; ntl < 2; ntl++)
#pragma unroll
            for (int r = 0; r < 8; r++) {
                int row = mt * 16 + r + mlo;
                int ch  = hv * HD + ntl * 16 + colL;
                outw[((size_t)win * NTOK + row) * CCH + ch] = f2bf(O[ntl][r]);
            }
    }
}

// ---------------- K4: proj GEMM (W in LDS via TDM) + reverse + residual ----------------
__global__ __launch_bounds__(256) void k_proj(const unsigned short* __restrict__ outw,
                                              const unsigned short* __restrict__ W,
                                              const float* __restrict__ bias,
                                              const float* __restrict__ x_in,
                                              float* __restrict__ x2) {
    __shared__ __align__(16) unsigned short s_w[96 * 96];    // 18432 B
    if ((threadIdx.x >> 5) == 0) {
        tdm_load_2d(lds_offset_of(s_w), W, 96, 96);
        __builtin_amdgcn_s_wait_tensorcnt(0);
    }
    __syncthreads();

    int mt   = (blockIdx.x * 256 + threadIdx.x) >> 5;
    int lane = threadIdx.x & 31;
    if (mt >= TOK / 16) return;
    v16bf A[3];
#pragma unroll
    for (int kc = 0; kc < 3; kc++) A[kc] = load_tile_bf(outw, mt * 16, CCH, kc * 32, lane);
    int colL = lane & 15, mhi = (lane >> 4) << 3;
#pragma unroll
    for (int nt = 0; nt < 6; nt++) {
        v8f acc = {};
#pragma unroll
        for (int kc = 0; kc < 3; kc++)
            acc = wmma_bf16(A[kc], load_tile_bf(s_w, nt * 16, CCH, kc * 32, lane), acc);
        int col = nt * 16 + colL;
        float bi = bias[col];
#pragma unroll
        for (int r = 0; r < 8; r++) {
            int o = mt * 16 + r + mhi;
            size_t a = win_to_global(o) * CCH + col;
            x2[a] = x_in[a] + acc[r] + bi;
        }
    }
}

// ---------------- K5: LN2 -> bf16 ----------------
__global__ __launch_bounds__(256) void k_ln2(const float* __restrict__ x2,
                                             const float* __restrict__ g,
                                             const float* __restrict__ b,
                                             unsigned short* __restrict__ ln) {
    int t    = (blockIdx.x * 256 + threadIdx.x) >> 5;
    int lane = threadIdx.x & 31;
    if (t >= TOK) return;
    const float* row = x2 + (size_t)t * CCH;
    float v0 = row[lane], v1 = row[lane + 32], v2 = row[lane + 64];
    float s = v0 + v1 + v2;
#pragma unroll
    for (int off = 16; off; off >>= 1) s += __shfl_xor(s, off);
    float mean = s * (1.0f / 96.0f);
    float d0 = v0 - mean, d1 = v1 - mean, d2 = v2 - mean;
    float q = d0 * d0 + d1 * d1 + d2 * d2;
#pragma unroll
    for (int off = 16; off; off >>= 1) q += __shfl_xor(q, off);
    float rstd = rsqrtf(q * (1.0f / 96.0f) + 1e-5f);
    unsigned short* out = ln + (size_t)t * CCH;
    out[lane]      = f2bf(d0 * rstd * g[lane]      + b[lane]);
    out[lane + 32] = f2bf(d1 * rstd * g[lane + 32] + b[lane + 32]);
    out[lane + 64] = f2bf(d2 * rstd * g[lane + 64] + b[lane + 64]);
}

// ---------------- K6: FC1 (96 -> 384) + exact GELU -> bf16 ----------------
__global__ __launch_bounds__(256) void k_fc1(const unsigned short* __restrict__ ln,
                                             const unsigned short* __restrict__ W,
                                             const float* __restrict__ bias,
                                             unsigned short* __restrict__ h1) {
    int mt   = (blockIdx.x * 256 + threadIdx.x) >> 5;
    int lane = threadIdx.x & 31;
    if (mt >= TOK / 16) return;
    __builtin_prefetch(W, 0, 3);
    v16bf A[3];
#pragma unroll
    for (int kc = 0; kc < 3; kc++) A[kc] = load_tile_bf(ln, mt * 16, CCH, kc * 32, lane);
    int colL = lane & 15, mhi = (lane >> 4) << 3;
#pragma unroll
    for (int nt = 0; nt < 24; nt++) {
        v8f acc = {};
#pragma unroll
        for (int kc = 0; kc < 3; kc++)
            acc = wmma_bf16(A[kc], load_tile_bf(W, nt * 16, CCH, kc * 32, lane), acc);
        int col = nt * 16 + colL;
        float bi = bias[col];
#pragma unroll
        for (int r = 0; r < 8; r++) {
            float v = acc[r] + bi;
            v = 0.5f * v * (1.0f + erff(v * 0.70710678118654752f));
            h1[(size_t)(mt * 16 + r + mhi) * 384 + col] = f2bf(v);
        }
    }
}

// ---------------- K7: FC2 (384 -> 96) + residual -> d_out fp32 ----------------
__global__ __launch_bounds__(256) void k_fc2(const unsigned short* __restrict__ h1,
                                             const unsigned short* __restrict__ W,
                                             const float* __restrict__ bias,
                                             const float* __restrict__ x2,
                                             float* __restrict__ out) {
    int mt   = (blockIdx.x * 256 + threadIdx.x) >> 5;
    int lane = threadIdx.x & 31;
    if (mt >= TOK / 16) return;
    __builtin_prefetch(W, 0, 3);
    v16bf A[12];
#pragma unroll
    for (int kc = 0; kc < 12; kc++) A[kc] = load_tile_bf(h1, mt * 16, 384, kc * 32, lane);
    int colL = lane & 15, mhi = (lane >> 4) << 3;
#pragma unroll
    for (int nt = 0; nt < 6; nt++) {
        v8f acc = {};
#pragma unroll
        for (int kc = 0; kc < 12; kc++)
            acc = wmma_bf16(A[kc], load_tile_bf(W, nt * 16, 384, kc * 32, lane), acc);
        int col = nt * 16 + colL;
        float bi = bias[col];
#pragma unroll
        for (int r = 0; r < 8; r++) {
            size_t a = (size_t)(mt * 16 + r + mhi) * CCH + col;
            out[a] = x2[a] + acc[r] + bi;
        }
    }
}

// ---------------- launcher ----------------
extern "C" void kernel_launch(void* const* d_in, const int* in_sizes, int n_in,
                              void* d_out, int out_size, void* d_ws, size_t ws_size,
                              hipStream_t stream) {
    (void)in_sizes; (void)n_in; (void)out_size; (void)ws_size;
    const float* x        = (const float*)d_in[0];
    const float* norm1_w  = (const float*)d_in[1];
    const float* norm1_b  = (const float*)d_in[2];
    const float* qkv_w    = (const float*)d_in[3];
    const float* qkv_b    = (const float*)d_in[4];
    const float* rel_bias = (const float*)d_in[5];
    const float* proj_w   = (const float*)d_in[6];
    const float* proj_b   = (const float*)d_in[7];
    const float* norm2_w  = (const float*)d_in[8];
    const float* norm2_b  = (const float*)d_in[9];
    const float* fc1_w    = (const float*)d_in[10];
    const float* fc1_b    = (const float*)d_in[11];
    const float* fc2_w    = (const float*)d_in[12];
    const float* fc2_b    = (const float*)d_in[13];
    float* out = (float*)d_out;

    char* ws = (char*)d_ws;
    const size_t OFF1 = (size_t)TOK * CCH * 2;                 // xw / ln2 (48 MiB)
    const size_t OFF2 = OFF1 + (size_t)TOK * 384 * 2;          // qkv / fc1 (192 MiB)
    const size_t OFF3 = OFF2 + (size_t)TOK * CCH * 2;          // attn out (48 MiB)
    const size_t OFF4 = OFF3 + (size_t)TOK * CCH * 4;          // x2 (96 MiB)
    unsigned short* xw   = (unsigned short*)ws;                // also ln2 output
    unsigned short* qkvb = (unsigned short*)(ws + OFF1);       // also fc1 output
    unsigned short* atto = (unsigned short*)(ws + OFF2);
    float*          x2   = (float*)(ws + OFF3);
    // bf16 weight copies (L2-resident)
    unsigned short* wqkv = (unsigned short*)(ws + OFF4);                    // 288*96
    unsigned short* wprj = wqkv + 288 * 96;                                 // 96*96
    unsigned short* wfc1 = wprj + 96 * 96;                                  // 384*96
    unsigned short* wfc2 = wfc1 + 384 * 96;                                 // 96*384

    k_cvt<<<(288 * 96 + 255) / 256, 256, 0, stream>>>(qkv_w, wqkv, 288 * 96);
    k_cvt<<<(96 * 96 + 255) / 256, 256, 0, stream>>>(proj_w, wprj, 96 * 96);
    k_cvt<<<(384 * 96 + 255) / 256, 256, 0, stream>>>(fc1_w, wfc1, 384 * 96);
    k_cvt<<<(96 * 384 + 255) / 256, 256, 0, stream>>>(fc2_w, wfc2, 96 * 384);

    k_ln1_win<<<TOK / 8, 256, 0, stream>>>(x, norm1_w, norm1_b, xw);
    k_qkv   <<<TOK / 16 / 8, 256, 0, stream>>>(xw, wqkv, qkv_b, qkvb);
    k_attn  <<<NWIN, 96, 0, stream>>>(qkvb, rel_bias, atto);
    k_proj  <<<TOK / 16 / 8, 256, 0, stream>>>(atto, wprj, proj_b, x, x2);
    k_ln2   <<<TOK / 8, 256, 0, stream>>>(x2, norm2_w, norm2_b, xw);
    k_fc1   <<<TOK / 16 / 8, 256, 0, stream>>>(xw, wfc1, fc1_b, qkvb);
    k_fc2   <<<TOK / 16 / 8, 256, 0, stream>>>(qkvb, wfc2, fc2_b, x2, out);
}
